// MultiHeadAttention_7301444403734
// MI455X (gfx1250) — matrix operand order repaired and checked by reading, not compiled, not run on hardware
//
#include <hip/hip_runtime.h>
#include <hip/hip_bf16.h>
#include <math.h>

// ---------------- problem constants ----------------
#define BB   2
#define SS   2048
#define DD   1024
#define HH   16
#define HD   64
#define MTOT (BB * SS)      // 4096 rows in all projection GEMMs

// ---------------- WMMA / TDM types ----------------
typedef __attribute__((ext_vector_type(16))) __bf16 v16bf;
typedef __attribute__((ext_vector_type(8)))  float  v8f;
typedef __attribute__((ext_vector_type(4)))  unsigned int v4u;
typedef __attribute__((ext_vector_type(8)))  int    v8i;
typedef __attribute__((ext_vector_type(4)))  int    v4i;

__device__ __forceinline__ v8f wmma_bf16(v16bf a, v16bf b, v8f c) {
  return __builtin_amdgcn_wmma_f32_16x16x32_bf16(
      /*neg_a=*/false, a, /*neg_b=*/false, b,
      /*c_mod=*/(short)0, c, /*reuse_a=*/false, /*reuse_b=*/false);
}

// f32 -> bf16 round-to-nearest-even
__device__ __forceinline__ unsigned short f2bf(float x) {
  unsigned int u = __float_as_uint(x);
  u += 0x7FFFu + ((u >> 16) & 1u);
  return (unsigned short)(u >> 16);
}

union FragU { uint4 u[2]; v16bf v; };

// A-matrix 16x32 bf16 fragment (ISA 7.12.2): lane L (r=L&15) holds row r,
// K elements {c0..c0+7, c0+16..c0+23} with c0 = 8*(L>=16). Row-major source.
__device__ __forceinline__ v16bf load_a16(const unsigned short* p, int ld) {
  const int lane = threadIdx.x & 31;
  const int r  = lane & 15;
  const int c0 = (lane >> 4) << 3;
  const unsigned short* q = p + (size_t)r * ld + c0;
  FragU f;
  f.u[0] = *(const uint4*)(q);
  f.u[1] = *(const uint4*)(q + 16);
  return f.v;
}

// B-matrix 32x16 bf16 fragment: lane L (n=L&15) holds column n of B,
// K elements {c0..c0+7, c0+16..c0+23}, c0 = 8*(L>=16). Source row-major B^T.
__device__ __forceinline__ v16bf load_b16(const unsigned short* p, int ld) {
  const int lane = threadIdx.x & 31;
  const int r  = lane & 15;
  const int c0 = (lane >> 4) << 3;
  const unsigned short* q = p + (size_t)r * ld + c0;
  FragU f;
  f.u[0] = *(const uint4*)(q);
  f.u[1] = *(const uint4*)(q + 16);
  return f.v;
}

// generic pointer -> 32-bit LDS offset (aperture low bits are the LDS addr)
__device__ __forceinline__ unsigned int lds_u32(const void* p) {
  return (unsigned int)(size_t)p;
}

// ---------------- Tensor Data Mover: 2D tile load (D# per ISA 8.3/8.4) ------
// tile_d0 elements along contiguous dim, tile_d1 rows, row stride stride_d0
// (all in 2-byte elements). pad codes: every 2^(pi+1) DWORDs insert (pa+1) DW.
__device__ __forceinline__ void tdm_load_2d(const void* gaddr, unsigned int lds_addr,
                                            unsigned int tile_d0, unsigned int tile_d1,
                                            unsigned int stride_d0,
                                            unsigned int pad_interval_code,
                                            unsigned int pad_amount_code) {
  unsigned long long ga = (unsigned long long)(size_t)gaddr;
  v4u g0;
  g0[0] = 1u;                                            // count=1, user mode
  g0[1] = lds_addr;                                      // lds_addr[31:0]
  g0[2] = (unsigned int)(ga & 0xFFFFFFFFull);            // global_addr[31:0]
  g0[3] = (unsigned int)((ga >> 32) & 0x01FFFFFFull)     // global_addr[56:32]
        | (2u << 30);                                    // type = 2 (image)
  v8i g1;
  unsigned int dw0 = (1u << 16)                          // data_size = 2 bytes
                   | (1u << 20)                          // pad_enable
                   | (pad_interval_code << 22)
                   | (pad_amount_code << 25);
  g1[0] = (int)dw0;
  g1[1] = (int)((tile_d0 & 0xFFFFu) << 16);              // tensor_dim0[15:0]
  g1[2] = (int)(((tile_d0 >> 16) & 0xFFFFu) |            // tensor_dim0[31:16]
                ((tile_d1 & 0xFFFFu) << 16));            // tensor_dim1[15:0]
  g1[3] = (int)(((tile_d1 >> 16) & 0xFFFFu) |            // tensor_dim1[31:16]
                ((tile_d0 & 0xFFFFu) << 16));            // tile_dim0
  g1[4] = (int)(tile_d1 & 0xFFFFu);                      // tile_dim1, tile_dim2=0
  g1[5] = (int)stride_d0;                                // tensor_dim0_stride[31:0]
  g1[6] = 0;                                             // stride hi, dim1_stride lo
  g1[7] = 0;
  v4i z4 = {0, 0, 0, 0};
#if __clang_major__ >= 23
  v8i z8 = {0, 0, 0, 0, 0, 0, 0, 0};
  __builtin_amdgcn_tensor_load_to_lds(g0, g1, z4, z4, z8, 0);
#else
  __builtin_amdgcn_tensor_load_to_lds(g0, g1, z4, z4, 0);
#endif
}

// ---------------- conversion kernels ----------------
__global__ void cvt_bf16(const float* __restrict__ x, unsigned short* __restrict__ y, int n) {
  int t = blockIdx.x * blockDim.x + threadIdx.x;
  if (t < n) y[t] = f2bf(x[t]);
}

__global__ void cvtT_bf16(const float* __restrict__ W, unsigned short* __restrict__ Wt) {
  int t = blockIdx.x * blockDim.x + threadIdx.x;   // DD*DD threads
  int k = t & (DD - 1);
  int n = t >> 10;
  Wt[(size_t)n * DD + k] = f2bf(W[(size_t)k * DD + n]);
}

// ---------------- GEMM: C = A[M,K]bf16 * Bt[N,K]bf16^T + bias ----------------
// All 8 waves of a block share mt -> A-chunks (32 x 128) are TDM-staged in LDS
// (double-buffered); B (weights) stays on the global path (L2-resident, 2 MB).
#define LDA 136   // 128 + 8 elements TDM pad -> conflict-free ds_load_b128
template <int MODE>
__global__ void gemm_wmma(const unsigned short* __restrict__ A,
                          const unsigned short* __restrict__ Bt,
                          const float* __restrict__ bias,
                          void* __restrict__ Cout,
                          int Mdim, int Ndim, int Kdim) {
  __shared__ unsigned short atile[2][32 * LDA];
  const int widx = threadIdx.x >> 5;
  const int lane = threadIdx.x & 31;
  const int wg = blockIdx.x * 8 + widx;
  const int ntiles = Ndim >> 5;
  const int nt = (wg % ntiles) << 5;
  const int mt = (wg / ntiles) << 5;   // uniform across the block's 8 waves

  v8f a00 = {}, a01 = {}, a10 = {}, a11 = {};
  const unsigned short* Ag  = A + (size_t)mt * Kdim;          // 32 rows
  const unsigned short* Bn0 = Bt + (size_t)nt * Kdim;
  const unsigned short* Bn1 = Bt + (size_t)(nt + 16) * Kdim;
  const unsigned int la0 = lds_u32(&atile[0][0]);
  const unsigned int la1 = lds_u32(&atile[1][0]);

  if (widx == 0) tdm_load_2d(Ag, la0, 128, 32, Kdim, 5, 3);

  const int nchunks = Kdim >> 7;                              // K in 128-chunks
  for (int c = 0; c < nchunks; ++c) {
    const int cur = c & 1;
    if (widx == 0) {
      if (c + 1 < nchunks) {
        tdm_load_2d(Ag + (size_t)(c + 1) * 128, cur ? la0 : la1, 128, 32, Kdim, 5, 3);
        __builtin_amdgcn_s_wait_tensorcnt(1);
      } else {
        __builtin_amdgcn_s_wait_tensorcnt(0);
      }
    }
    __syncthreads();
    const unsigned short* at = atile[cur];
    const int kbase = c << 7;
#pragma unroll
    for (int kk = 0; kk < 4; ++kk) {
      const int k0 = kbase + kk * 32;
      __builtin_prefetch(Bn0 + k0 + 256, 0, 3);
      __builtin_prefetch(Bn1 + k0 + 256, 0, 3);
      v16bf fa0 = load_a16(at + kk * 32, LDA);
      v16bf fa1 = load_a16(at + 16 * LDA + kk * 32, LDA);
      v16bf fb0 = load_b16(Bn0 + k0, Kdim);
      v16bf fb1 = load_b16(Bn1 + k0, Kdim);
      a00 = wmma_bf16(fa0, fb0, a00);
      a01 = wmma_bf16(fa0, fb1, a01);
      a10 = wmma_bf16(fa1, fb0, a10);
      a11 = wmma_bf16(fa1, fb1, a11);
    }
    __syncthreads();
  }

  const int col = lane & 15;
  const int rb  = (lane >> 4) << 3;

  auto store_frag = [&](v8f c, int mbase, int nbase) {
    float bv = bias[nbase + col];
    if constexpr (MODE == 2) {
      float* C = (float*)Cout;
      for (int i = 0; i < 8; ++i)
        C[(size_t)(mbase + rb + i) * Ndim + nbase + col] = c[i] + bv;
    } else if constexpr (MODE == 1) {
      unsigned short* C = (unsigned short*)Cout;
      union { uint4 u; unsigned short s[8]; } pk;
      for (int i = 0; i < 8; ++i) pk.s[i] = f2bf(c[i] + bv);
      *(uint4*)(C + (size_t)(nbase + col) * Mdim + mbase + rb) = pk.u;
    } else {
      unsigned short* C = (unsigned short*)Cout;
      for (int i = 0; i < 8; ++i)
        C[(size_t)(mbase + rb + i) * Ndim + nbase + col] = f2bf(c[i] + bv);
    }
  };
  store_frag(a00, mt, nt);
  store_frag(a01, mt, nt + 16);
  store_frag(a10, mt + 16, nt);
  store_frag(a11, mt + 16, nt + 16);
}

// ---------------- pass 1: per-key column stats (softmax over q axis) --------
#define LDQ 72   // 64 + 8 elements of TDM pad -> conflict-free ds reads
__global__ void attn_colstats(const unsigned short* __restrict__ qp,
                              const unsigned short* __restrict__ kp,
                              float* __restrict__ colmax,
                              float* __restrict__ colsum) {
  __shared__ unsigned short qtile[2][16 * LDQ];
  const int widx = threadIdx.x >> 5;
  const int lane = threadIdx.x & 31;
  const int bid = blockIdx.x;
  const int kblk = bid & 15;         // SS/128 = 16
  const int h = (bid >> 4) & 15;
  const int b = bid >> 8;
  const int kb = kblk * 128 + widx * 16;

  const unsigned short* kbase = kp + (size_t)(b * SS + kb) * DD + h * HD;
  const v16bf bk0 = load_b16(kbase, DD);        // d = 0..31
  const v16bf bk1 = load_b16(kbase + 32, DD);   // d = 32..63

  const unsigned short* qg = qp + (size_t)(b * SS) * DD + h * HD;
  const unsigned int lq0 = lds_u32(&qtile[0][0]);
  const unsigned int lq1 = lds_u32(&qtile[1][0]);

  if (widx == 0) tdm_load_2d(qg, lq0, 64, 16, DD, 4, 3);

  float m = -INFINITY, sum = 0.f;
  for (int it = 0; it < SS / 16; ++it) {
    const int cur = it & 1;
    if (widx == 0) {
      if (it + 1 < SS / 16) {
        tdm_load_2d(qg + (size_t)(it + 1) * 16 * DD, cur ? lq0 : lq1, 64, 16, DD, 4, 3);
        __builtin_amdgcn_s_wait_tensorcnt(1);
      } else {
        __builtin_amdgcn_s_wait_tensorcnt(0);
      }
    }
    __syncthreads();
    const unsigned short* qt_ = qtile[cur];
    v16bf a0 = load_a16(qt_, LDQ);
    v16bf a1 = load_a16(qt_ + 32, LDQ);
    v8f c = {};
    c = wmma_bf16(a0, bk0, c);
    c = wmma_bf16(a1, bk1, c);
    float mn = m;
    for (int i = 0; i < 8; ++i) mn = fmaxf(mn, c[i]);
    sum *= __expf(m - mn);
    for (int i = 0; i < 8; ++i) sum += __expf(c[i] - mn);
    m = mn;
    __syncthreads();
  }
  float m2 = __shfl_xor(m, 16, 32);
  float s2 = __shfl_xor(sum, 16, 32);
  float Mx = fmaxf(m, m2);
  float Sx = sum * __expf(m - Mx) + s2 * __expf(m2 - Mx);
  if (lane < 16) {
    int idx = (b * HH + h) * SS + kb + lane;
    colmax[idx] = Mx;
    colsum[idx] = Sx;
  }
}

// ---------------- pass 2: out = softmax_cols(QK^T) * V ----------------------
#define LDP 40
#define LDK 72
#define LDV 40
__global__ void attn_out_kernel(const unsigned short* __restrict__ qp,
                                const unsigned short* __restrict__ kp,
                                const unsigned short* __restrict__ vpT,
                                const float* __restrict__ colmax,
                                const float* __restrict__ colsum,
                                unsigned short* __restrict__ attnb) {
  __shared__ unsigned short ktile[2][32 * LDK];
  __shared__ unsigned short vtile[2][64 * LDV];
  __shared__ unsigned short ptile[8][16 * LDP];
  const int widx = threadIdx.x >> 5;
  const int lane = threadIdx.x & 31;

  const int wg = blockIdx.x * 8 + widx;
  const int qt = (wg & 127) * 16;
  const int h  = (wg >> 7) & 15;
  const int b  = wg >> 11;

  const unsigned short* qbase = qp + (size_t)(b * SS + qt) * DD + h * HD;
  const v16bf a0 = load_a16(qbase, DD);
  const v16bf a1 = load_a16(qbase + 32, DD);

  const int col = lane & 15;
  const int rb  = (lane >> 4) << 3;
  const float* cm = colmax + (size_t)(b * HH + h) * SS;
  const float* cs = colsum + (size_t)(b * HH + h) * SS;
  const unsigned short* kg = kp + (size_t)(b * SS) * DD + h * HD;    // rows=keys
  const unsigned short* vg = vpT + (size_t)(h * HD) * MTOT + b * SS; // rows=d
  unsigned short* myP = ptile[widx];

  const unsigned int lk0 = lds_u32(&ktile[0][0]), lk1 = lds_u32(&ktile[1][0]);
  const unsigned int lv0 = lds_u32(&vtile[0][0]), lv1 = lds_u32(&vtile[1][0]);

  if (widx == 0) {
    tdm_load_2d(kg, lk0, 64, 32, DD, 4, 3);    // K: 32 keys x 64 d
    tdm_load_2d(vg, lv0, 32, 64, MTOT, 3, 3);  // V^T: 64 d x 32 keys
  }

  v8f o0 = {}, o1 = {}, o2 = {}, o3 = {};
  for (int it = 0; it < SS / 32; ++it) {
    const int cur = it & 1;
    if (widx == 0) {
      if (it + 1 < SS / 32) {
        tdm_load_2d(kg + (size_t)(it + 1) * 32 * DD, cur ? lk0 : lk1, 64, 32, DD, 4, 3);
        tdm_load_2d(vg + (size_t)(it + 1) * 32,      cur ? lv0 : lv1, 32, 64, MTOT, 3, 3);
        __builtin_amdgcn_s_wait_tensorcnt(2);   // current pair complete (in-order)
      } else {
        __builtin_amdgcn_s_wait_tensorcnt(0);
      }
    }
    __syncthreads();

    const unsigned short* kt = ktile[cur];
    const unsigned short* vt = vtile[cur];

    v8f s0 = {}, s1 = {};
    s0 = wmma_bf16(a0, load_b16(kt, LDK), s0);
    s0 = wmma_bf16(a1, load_b16(kt + 32, LDK), s0);
    s1 = wmma_bf16(a0, load_b16(kt + 16 * LDK, LDK), s1);
    s1 = wmma_bf16(a1, load_b16(kt + 16 * LDK + 32, LDK), s1);

    const int kb = it * 32;
    const float m0v = cm[kb + col];
    const float i0v = __frcp_rn(cs[kb + col]);
    const float m1v = cm[kb + 16 + col];
    const float i1v = __frcp_rn(cs[kb + 16 + col]);

    for (int i = 0; i < 8; ++i) {
      myP[(rb + i) * LDP + col]      = f2bf(__expf(s0[i] - m0v) * i0v);
      myP[(rb + i) * LDP + 16 + col] = f2bf(__expf(s1[i] - m1v) * i1v);
    }
    asm volatile("s_wait_dscnt 0" ::: "memory");
    __builtin_amdgcn_wave_barrier();

    v16bf pa = load_a16(myP, LDP);   // P 16x32 A-fragment
    o0 = wmma_bf16(pa, load_b16(vt + 0 * 16 * LDV, LDV), o0);
    o1 = wmma_bf16(pa, load_b16(vt + 1 * 16 * LDV, LDV), o1);
    o2 = wmma_bf16(pa, load_b16(vt + 2 * 16 * LDV, LDV), o2);
    o3 = wmma_bf16(pa, load_b16(vt + 3 * 16 * LDV, LDV), o3);
    __syncthreads();
  }

  unsigned short* obase = attnb + (size_t)(b * SS + qt) * DD + h * HD;
  for (int i = 0; i < 8; ++i) {
    obase[(size_t)(rb + i) * DD + col]      = f2bf(o0[i]);
    obase[(size_t)(rb + i) * DD + 16 + col] = f2bf(o1[i]);
    obase[(size_t)(rb + i) * DD + 32 + col] = f2bf(o2[i]);
    obase[(size_t)(rb + i) * DD + 48 + col] = f2bf(o3[i]);
  }
}

// ---------------- host-side orchestration ----------------
extern "C" void kernel_launch(void* const* d_in, const int* in_sizes, int n_in,
                              void* d_out, int out_size, void* d_ws, size_t ws_size,
                              hipStream_t stream) {
  const float* q  = (const float*)d_in[0];
  const float* k  = (const float*)d_in[1];
  const float* v  = (const float*)d_in[2];
  const float* Wq = (const float*)d_in[3];
  const float* bq = (const float*)d_in[4];
  const float* Wk = (const float*)d_in[5];
  const float* bk = (const float*)d_in[6];
  const float* Wv = (const float*)d_in[7];
  const float* bv = (const float*)d_in[8];
  const float* Wo = (const float*)d_in[9];
  const float* bo = (const float*)d_in[10];

  const size_t MD = (size_t)MTOT * DD;           // 4M elements
  char* w = (char*)d_ws;
  size_t off = 0;
  auto alloc = [&](size_t bytes) { void* p = w + off; off += (bytes + 255) & ~(size_t)255; return p; };

  unsigned short* qbf  = (unsigned short*)alloc(MD * 2);
  unsigned short* kbf  = (unsigned short*)alloc(MD * 2);
  unsigned short* vbf  = (unsigned short*)alloc(MD * 2);
  unsigned short* WqT  = (unsigned short*)alloc((size_t)DD * DD * 2);
  unsigned short* WkT  = (unsigned short*)alloc((size_t)DD * DD * 2);
  unsigned short* WvT  = (unsigned short*)alloc((size_t)DD * DD * 2);
  unsigned short* WoT  = (unsigned short*)alloc((size_t)DD * DD * 2);
  unsigned short* qpb  = (unsigned short*)alloc(MD * 2);
  unsigned short* kpb  = (unsigned short*)alloc(MD * 2);
  unsigned short* vpT  = (unsigned short*)alloc(MD * 2);   // [DD][MTOT]
  unsigned short* attb = (unsigned short*)alloc(MD * 2);
  float* colmax = (float*)alloc((size_t)BB * HH * SS * 4);
  float* colsum = (float*)alloc((size_t)BB * HH * SS * 4);

  // 1) fp32 -> bf16 activations, fp32 -> bf16 transposed weights
  cvt_bf16<<<(int)(MD / 256), 256, 0, stream>>>(q, qbf, (int)MD);
  cvt_bf16<<<(int)(MD / 256), 256, 0, stream>>>(k, kbf, (int)MD);
  cvt_bf16<<<(int)(MD / 256), 256, 0, stream>>>(v, vbf, (int)MD);
  cvtT_bf16<<<(DD * DD) / 256, 256, 0, stream>>>(Wq, WqT);
  cvtT_bf16<<<(DD * DD) / 256, 256, 0, stream>>>(Wk, WkT);
  cvtT_bf16<<<(DD * DD) / 256, 256, 0, stream>>>(Wv, WvT);
  cvtT_bf16<<<(DD * DD) / 256, 256, 0, stream>>>(Wo, WoT);

  // 2) projections (512 blocks * 8 waves = 4096 32x32 tiles)
  gemm_wmma<0><<<512, 256, 0, stream>>>(qbf, WqT, bq, qpb, MTOT, DD, DD);
  gemm_wmma<0><<<512, 256, 0, stream>>>(kbf, WkT, bk, kpb, MTOT, DD, DD);
  gemm_wmma<1><<<512, 256, 0, stream>>>(vbf, WvT, bv, vpT, MTOT, DD, DD); // V^T

  // 3) column softmax stats, then attention output
  attn_colstats<<<BB * HH * (SS / 128), 256, 0, stream>>>(qpb, kpb, colmax, colsum);
  attn_out_kernel<<<BB * HH * (SS / (16 * 8)), 256, 0, stream>>>(
      qpb, kpb, vpT, colmax, colsum, attb);

  // 4) output projection to f32 d_out
  gemm_wmma<2><<<512, 256, 0, stream>>>(attb, WoT, bo, d_out, MTOT, DD, DD);
}
